// TransitionModel_20375324852595
// MI455X (gfx1250) — compile-verified, matrix operand order checked by reading, not run on hardware
//
#include <hip/hip_runtime.h>
#include <math.h>

// ---------------------------------------------------------------------------
// Problem dimensions (TransitionModel)
// ---------------------------------------------------------------------------
#define B_   4096
#define L_   1024
#define BEL_ 2048
#define H_   2048
#define A_   512
#define OBS_ 2048
#define OUTW (4 * L_ + BEL_)   // 6144

// WMMA tile config
#define BM 128
#define BN 128
#define BK 32

typedef __attribute__((ext_vector_type(16))) __bf16 v16bf;
typedef __attribute__((ext_vector_type(8)))  float  v8f;
typedef __attribute__((ext_vector_type(2)))  float  v2f;
typedef __attribute__((ext_vector_type(2)))  __bf16 v2bf;

union FragBF16 {
    unsigned u[8];
    v16bf    v;
};

// One v_cvt_pk_bf16_f32: two floats -> packed bf16x2 in a single VGPR.
__device__ __forceinline__ unsigned cvt_pk2(float x, float y) {
    v2f f;
    f.x = x;
    f.y = y;
    v2bf b = __builtin_convertvector(f, v2bf);
    return __builtin_bit_cast(unsigned, b);
}

// ACT: 0 = none, 1 = ELU, 2 = softplus(x)+0.1
template <int ACT>
__device__ __forceinline__ float act_fn(float x) {
    if (ACT == 1) {
        return x > 0.f ? x : (expf(x) - 1.f);
    } else if (ACT == 2) {
        float sp = (x > 20.f) ? x : log1pf(expf(x));
        return sp + 0.1f;
    }
    return x;
}

// ---------------------------------------------------------------------------
// Fused GEMM + bias + activation:  C[:, col_off:col_off+N] = act(A @ W + b)
//   A: [M, K]  fp32 row-major (contiguous, lda = K)
//   W: [K, N]  fp32 row-major
//   b: [N]     fp32
//   C: fp32, row stride ldc, written at column offset col_off
// Requires: M % 128 == 0, N % 128 == 0, K % 32 == 0. All buffers < 4GB so
// 32-bit divergent offsets + uniform SGPR base (GVS addressing) are used.
// 256 threads = 8 waves; each wave computes a 32x64 sub-tile via
// 2x4 = 8 v_wmma_f32_16x16x32_bf16 per BK=32 K-step.
// ---------------------------------------------------------------------------
template <int ACT>
__global__ __launch_bounds__(256)
void gemm_bias_act(const float* __restrict__ A,
                   const float* __restrict__ W,
                   const float* __restrict__ bias,
                   float* __restrict__ C,
                   int M, int N, int K, int ldc, int col_off)
{
    // A tile [row][k] row-major (K contiguous); B tile [col][k] transposed.
    // Stored as uint arrays (2 bf16 per uint) for packed access.
    __shared__ __align__(16) unsigned As32[BM * BK / 2];   // 2048 uints
    __shared__ __align__(16) unsigned Bs32[BN * BK / 2];   // 2048 uints

    const int tid    = threadIdx.x;
    const int lane   = tid & 31;
    const int wid    = tid >> 5;     // 0..7
    const int wr     = wid >> 1;     // 0..3  -> 32-row band
    const int wc     = wid & 1;      // 0..1  -> 64-col band
    const int laneLo = lane & 15;
    const int kb2    = (lane >> 4) << 2;  // uint-pair offset: 0 (lanes 0-15) or 4 (lanes 16-31)

    const int rowBase = blockIdx.y * BM;
    const int colBase = blockIdx.x * BN;

    // Loop-invariant 32-bit element offsets (divergent part of addresses).
    unsigned aIdx[4], bIdx[4];
#pragma unroll
    for (int i = 0; i < 4; ++i) {
        int e4 = tid + i * 256;       // float4 index in 128x32 A tile
        int r  = e4 >> 3;
        int kq = e4 & 7;
        aIdx[i] = (unsigned)(rowBase + r) * (unsigned)K + (unsigned)(kq * 4);
    }
#pragma unroll
    for (int i = 0; i < 4; ++i) {
        int c  = tid + i * 256;       // 4(k) x 1(n) chunk in 32x128 B tile
        int n  = c & 127;
        int kq = c >> 7;
        bIdx[i] = (unsigned)(kq * 4) * (unsigned)N + (unsigned)(colBase + n);
    }

    v8f acc[2][4];
    const v8f vzero = {};
#pragma unroll
    for (int mi = 0; mi < 2; ++mi)
#pragma unroll
        for (int ni = 0; ni < 4; ++ni)
            acc[mi][ni] = vzero;

    for (int k0 = 0; k0 < K; k0 += BK) {
        const float* Ak = A + k0;               // uniform (SGPR) base advance
        const float* Wk = W + (size_t)k0 * N;   // uniform (SGPR) base advance

        // ---- phase 1: issue all global loads for this K-step as a clause ----
        float4 av[4];
        float  wv[4][4];
#pragma unroll
        for (int i = 0; i < 4; ++i)
            av[i] = *reinterpret_cast<const float4*>(Ak + aIdx[i]);
#pragma unroll
        for (int i = 0; i < 4; ++i) {
#pragma unroll
            for (int j = 0; j < 4; ++j)
                wv[i][j] = Wk[bIdx[i] + (unsigned)(j * N)];  // j*N -> ioffset
        }
        // prefetch next K tile into cache (global_prefetch_b8)
        if (k0 + BK < K) {
            __builtin_prefetch(Ak + BK + aIdx[0], 0, 1);
            __builtin_prefetch(Wk + (size_t)BK * N + bIdx[0], 0, 1);
        }

        // ---- phase 2: convert (v_cvt_pk_bf16_f32) + LDS b64 stores ----
#pragma unroll
        for (int i = 0; i < 4; ++i) {
            int e4 = tid + i * 256;
            int r  = e4 >> 3;
            int kq = e4 & 7;
            uint2 p;
            p.x = cvt_pk2(av[i].x, av[i].y);
            p.y = cvt_pk2(av[i].z, av[i].w);
            reinterpret_cast<uint2*>(As32)[r * 8 + kq] = p;
        }
#pragma unroll
        for (int i = 0; i < 4; ++i) {
            int c  = tid + i * 256;
            int n  = c & 127;
            int kq = c >> 7;
            uint2 p;
            p.x = cvt_pk2(wv[i][0], wv[i][1]);
            p.y = cvt_pk2(wv[i][2], wv[i][3]);
            reinterpret_cast<uint2*>(Bs32)[n * 8 + kq] = p;
        }
        __syncthreads();

        // ---- build fragments per ISA 16-bit layouts ----
        // A 16x32: lane l<16 holds row M=l, K{0..7,16..23}; lane l+16 holds K{8..15,24..31}
        FragBF16 af[2], bf[4];
#pragma unroll
        for (int mi = 0; mi < 2; ++mi) {
            int r = wr * 32 + mi * 16 + laneLo;
#pragma unroll
            for (int v = 0; v < 8; ++v)
                af[mi].u[v] = As32[r * 16 + kb2 + v + ((v >= 4) ? 4 : 0)];
        }
#pragma unroll
        for (int ni = 0; ni < 4; ++ni) {
            int c = wc * 64 + ni * 16 + laneLo;
#pragma unroll
            for (int v = 0; v < 8; ++v)
                bf[ni].u[v] = Bs32[c * 16 + kb2 + v + ((v >= 4) ? 4 : 0)];
        }

        // ---- 8 WMMAs per K-step ----
#pragma unroll
        for (int mi = 0; mi < 2; ++mi)
#pragma unroll
            for (int ni = 0; ni < 4; ++ni)
                acc[mi][ni] = __builtin_amdgcn_wmma_f32_16x16x32_bf16(
                    /*neg_a=*/false, af[mi].v,
                    /*neg_b=*/false, bf[ni].v,
                    /*c_mod=*/(short)0, acc[mi][ni],
                    /*reuse_a=*/false, /*reuse_b=*/false);

        __syncthreads();
    }

    // ---- epilogue: bias + activation, C f32 layout (VGPR r: M=r / M=8+r) ----
    const int rHi = (lane >= 16) ? 8 : 0;
#pragma unroll
    for (int mi = 0; mi < 2; ++mi) {
#pragma unroll
        for (int ni = 0; ni < 4; ++ni) {
            int colG = colBase + wc * 64 + ni * 16 + laneLo;
            float bv = bias[colG];
#pragma unroll
            for (int r = 0; r < 8; ++r) {
                int rowG = rowBase + wr * 32 + mi * 16 + r + rHi;
                unsigned coff = (unsigned)rowG * (unsigned)ldc +
                                (unsigned)(col_off + colG);
                float x = acc[mi][ni][r] + bv;
                C[coff] = act_fn<ACT>(x);
            }
        }
    }
}

// ---------------------------------------------------------------------------
// Row-wise concat (float4 granularity): out[r, :] = [a[r, :na] | b[r, :nb]]
// na, nb must be multiples of 4.
// ---------------------------------------------------------------------------
__global__ void concat2_kernel(const float* __restrict__ a, int na,
                               const float* __restrict__ b, int nb,
                               float* __restrict__ out, int rows)
{
    int na4 = na >> 2, nb4 = nb >> 2;
    int nc4 = na4 + nb4;
    long total = (long)rows * nc4;
    const float4* a4 = reinterpret_cast<const float4*>(a);
    const float4* b4 = reinterpret_cast<const float4*>(b);
    float4* o4 = reinterpret_cast<float4*>(out);
    for (long i = blockIdx.x * (long)blockDim.x + threadIdx.x; i < total;
         i += (long)gridDim.x * blockDim.x) {
        unsigned row = (unsigned)(i / nc4);
        unsigned col = (unsigned)(i % nc4);
        o4[(unsigned)i] = (col < (unsigned)na4)
                              ? a4[row * (unsigned)na4 + col]
                              : b4[row * (unsigned)nb4 + (col - (unsigned)na4)];
    }
}

// ---------------------------------------------------------------------------
// GRU combine (PyTorch convention); writes belief to scratch AND to the
// output's column band [4L, 4L+BEL).
// ---------------------------------------------------------------------------
__global__ void gru_kernel(const float* __restrict__ gi,
                           const float* __restrict__ gh,
                           const float* __restrict__ state,
                           float* __restrict__ belief,
                           float* __restrict__ out)
{
    long total = (long)B_ * BEL_;
    for (long i = blockIdx.x * (long)blockDim.x + threadIdx.x; i < total;
         i += (long)gridDim.x * blockDim.x) {
        unsigned row = (unsigned)(i / BEL_);
        unsigned c   = (unsigned)(i % BEL_);
        unsigned base = row * (3u * BEL_);
        float i_r = gi[base + c];
        float i_z = gi[base + BEL_ + c];
        float i_n = gi[base + 2u * BEL_ + c];
        float h_r = gh[base + c];
        float h_z = gh[base + BEL_ + c];
        float h_n = gh[base + 2u * BEL_ + c];
        float r = 1.f / (1.f + expf(-(i_r + h_r)));
        float z = 1.f / (1.f + expf(-(i_z + h_z)));
        float n = tanhf(i_n + r * h_n);
        float s = state[row * (unsigned)BEL_ + c];
        float bel = (1.f - z) * n + z * s;
        belief[row * (unsigned)BEL_ + c] = bel;
        out[row * (unsigned)OUTW + 4u * L_ + c] = bel;
    }
}

// ---------------------------------------------------------------------------
// Launch sequence
// ---------------------------------------------------------------------------
extern "C" void kernel_launch(void* const* d_in, const int* in_sizes, int n_in,
                              void* d_out, int out_size, void* d_ws, size_t ws_size,
                              hipStream_t stream)
{
    const float* prev_sample = (const float*)d_in[0];
    const float* prev_action = (const float*)d_in[1];
    const float* rnn_state   = (const float*)d_in[2];
    const float* obs         = (const float*)d_in[3];
    const float* t0_W = (const float*)d_in[4];  const float* t0_b = (const float*)d_in[5];
    const float* t1_W = (const float*)d_in[6];  const float* t1_b = (const float*)d_in[7];
    const float* W_ih = (const float*)d_in[8];  const float* W_hh = (const float*)d_in[9];
    const float* b_ih = (const float*)d_in[10]; const float* b_hh = (const float*)d_in[11];
    const float* p0_W = (const float*)d_in[12]; const float* p0_b = (const float*)d_in[13];
    const float* p1_W = (const float*)d_in[14]; const float* p1_b = (const float*)d_in[15];
    const float* pm_W = (const float*)d_in[16]; const float* pm_b = (const float*)d_in[17];
    const float* ps_W = (const float*)d_in[18]; const float* ps_b = (const float*)d_in[19];
    const float* q0_W = (const float*)d_in[20]; const float* q0_b = (const float*)d_in[21];
    const float* q1_W = (const float*)d_in[22]; const float* q1_b = (const float*)d_in[23];
    const float* qm_W = (const float*)d_in[24]; const float* qm_b = (const float*)d_in[25];
    const float* qs_W = (const float*)d_in[26]; const float* qs_b = (const float*)d_in[27];

    float* out = (float*)d_out;
    float* ws  = (float*)d_ws;

    // scratch layout (floats)
    size_t off = 0;
    float* cat    = ws + off; off += (size_t)B_ * (BEL_ + OBS_);  // max concat
    float* buf1   = ws + off; off += (size_t)B_ * H_;
    float* buf2   = ws + off; off += (size_t)B_ * H_;
    float* gi     = ws + off; off += (size_t)B_ * 3 * BEL_;
    float* gh     = ws + off; off += (size_t)B_ * 3 * BEL_;
    float* belief = ws + off; off += (size_t)B_ * BEL_;
    (void)ws_size; (void)in_sizes; (void)n_in; (void)out_size;

    const dim3 blk(256);
    auto ggrid = [](int M, int N) { return dim3(N / BN, M / BM); };

    // --- _transition MLP ---
    concat2_kernel<<<2048, blk, 0, stream>>>(prev_sample, L_, prev_action, A_, cat, B_);
    gemm_bias_act<1><<<ggrid(B_, H_), blk, 0, stream>>>(cat, t0_W, t0_b, buf1,
                                                        B_, H_, L_ + A_, H_, 0);
    gemm_bias_act<1><<<ggrid(B_, H_), blk, 0, stream>>>(buf1, t1_W, t1_b, buf2,
                                                        B_, H_, H_, H_, 0);

    // --- GRU cell ---
    gemm_bias_act<0><<<ggrid(B_, 3 * BEL_), blk, 0, stream>>>(buf2, W_ih, b_ih, gi,
                                                              B_, 3 * BEL_, H_, 3 * BEL_, 0);
    gemm_bias_act<0><<<ggrid(B_, 3 * BEL_), blk, 0, stream>>>(rnn_state, W_hh, b_hh, gh,
                                                              B_, 3 * BEL_, BEL_, 3 * BEL_, 0);
    gru_kernel<<<4096, blk, 0, stream>>>(gi, gh, rnn_state, belief, out);

    // --- prior head ---
    gemm_bias_act<1><<<ggrid(B_, H_), blk, 0, stream>>>(belief, p0_W, p0_b, buf1,
                                                        B_, H_, BEL_, H_, 0);
    gemm_bias_act<1><<<ggrid(B_, H_), blk, 0, stream>>>(buf1, p1_W, p1_b, buf2,
                                                        B_, H_, H_, H_, 0);
    gemm_bias_act<0><<<ggrid(B_, L_), blk, 0, stream>>>(buf2, pm_W, pm_b, out,
                                                        B_, L_, H_, OUTW, 0);
    gemm_bias_act<2><<<ggrid(B_, L_), blk, 0, stream>>>(buf2, ps_W, ps_b, out,
                                                        B_, L_, H_, OUTW, L_);

    // --- posterior head ---
    concat2_kernel<<<2048, blk, 0, stream>>>(belief, BEL_, obs, OBS_, cat, B_);
    gemm_bias_act<1><<<ggrid(B_, H_), blk, 0, stream>>>(cat, q0_W, q0_b, buf1,
                                                        B_, H_, BEL_ + OBS_, H_, 0);
    gemm_bias_act<1><<<ggrid(B_, H_), blk, 0, stream>>>(buf1, q1_W, q1_b, buf2,
                                                        B_, H_, H_, H_, 0);
    gemm_bias_act<0><<<ggrid(B_, L_), blk, 0, stream>>>(buf2, qm_W, qm_b, out,
                                                        B_, L_, H_, OUTW, 2 * L_);
    gemm_bias_act<2><<<ggrid(B_, L_), blk, 0, stream>>>(buf2, qs_W, qs_b, out,
                                                        B_, L_, H_, OUTW, 3 * L_);
}
